// MultiHeadMPSRecurrence_49392123904228
// MI455X (gfx1250) — matrix operand order repaired and checked by reading, not compiled, not run on hardware
//
#include <hip/hip_runtime.h>
#include <hip/hip_bf16.h>
#include <math.h>

#define BB 4
#define TT 8192
#define DM 1024
#define DH 1024
#define MTOT (BB*TT)      // 32768 rows
#define NC 32             // scan chunks
#define CL (TT/NC)        // 256 steps per chunk
#define LN_EPS 1e-5f

typedef __bf16 bf16;
typedef __attribute__((ext_vector_type(16))) __bf16 v16bf;
typedef __attribute__((ext_vector_type(8)))  __bf16 v8bf;
typedef __attribute__((ext_vector_type(4)))  __bf16 v4bf;
typedef __attribute__((ext_vector_type(8)))  float  v8f;
typedef __attribute__((ext_vector_type(4)))  float  f32x4;
typedef __attribute__((ext_vector_type(4)))  unsigned int u32x4;
typedef __attribute__((ext_vector_type(4)))  int i32x4;

// ---------- CDNA5 async global->LDS (ASYNCcnt) with portable fallback ----------
#if defined(__has_builtin)
#  if __has_builtin(__builtin_amdgcn_global_load_async_to_lds_b128)
#    define HAS_ASYNC_LDS 1
#  endif
#endif
#ifndef HAS_ASYNC_LDS
#  define HAS_ASYNC_LDS 0
#endif

typedef __attribute__((address_space(1))) i32x4* g_v4i;
typedef __attribute__((address_space(3))) i32x4* l_v4i;

__device__ __forceinline__ void cp16_g2l(const bf16* g, bf16* l) {
#if HAS_ASYNC_LDS
  __builtin_amdgcn_global_load_async_to_lds_b128((g_v4i)g, (l_v4i)l, 0, 0);
#else
  *(u32x4*)l = *(const u32x4*)g;
#endif
}

__device__ __forceinline__ void async_wait_all() {
#if HAS_ASYNC_LDS
#  if defined(__has_builtin) && __has_builtin(__builtin_amdgcn_s_wait_asynccnt)
  __builtin_amdgcn_s_wait_asynccnt(0);
#  else
  asm volatile("s_wait_asynccnt 0x0" ::: "memory");
#  endif
#endif
}

// ---------- WMMA fragment helpers ----------
union Frag16 { v16bf v; v8bf h[2]; };

// A fragment, 16x32 bf16 per ISA layout:
// lane<16: M=lane, K=0..7 (v0-3) and 16..23 (v4-7); lane>=16: M=lane-16, K=8..15 & 24..31
__device__ __forceinline__ v16bf frag_a(const bf16* sm, int lda, int m, int lane) {
  const int k0 = (lane < 16) ? 0 : 8;
  Frag16 f;
  f.h[0] = *(const v8bf*)(sm + m * lda + k0);
  f.h[1] = *(const v8bf*)(sm + m * lda + k0 + 16);
  return f.v;
}
// B fragment, 32x16 bf16: lane<16: N=lane, K=0..7 & 16..23; lane>=16: N=lane-16, K=8..15 & 24..31.
// BT tile stored row-major [N][K] so each half is 8 contiguous elements.
__device__ __forceinline__ v16bf frag_b(const bf16* sm, int ldb, int n, int lane) {
  const int kb = (lane < 16) ? 0 : 8;
  Frag16 f;
  f.h[0] = *(const v8bf*)(sm + n * ldb + kb);
  f.h[1] = *(const v8bf*)(sm + n * ldb + kb + 16);
  return f.v;
}

#define WMMA_BF16(a, b, c) \
  __builtin_amdgcn_wmma_f32_16x16x32_bf16(false, (a), false, (b), (short)0, (c), false, false)

// ---------------- conversion / transpose ----------------
__global__ __launch_bounds__(256) void k_cvt(const float* __restrict__ x,
                                             bf16* __restrict__ xb) {
  const size_t i = ((size_t)blockIdx.x * 256 + threadIdx.x) * 4;
  f32x4 v = *(const f32x4*)(x + i);
  *(v4bf*)(xb + i) = __builtin_convertvector(v, v4bf);
}

// w: [K][N] fp32 row-major -> wT: [N][K] bf16 row-major
__global__ __launch_bounds__(256) void k_transpose(const float* __restrict__ w,
                                                   bf16* __restrict__ wT,
                                                   int K, int N) {
  __shared__ float tile[32][33];
  const int bx = blockIdx.x * 32;  // n
  const int by = blockIdx.y * 32;  // k
  const int tx = threadIdx.x, ty = threadIdx.y;  // (32,8)
  #pragma unroll
  for (int j = 0; j < 32; j += 8)
    tile[ty + j][tx] = w[(size_t)(by + ty + j) * N + bx + tx];
  __syncthreads();
  #pragma unroll
  for (int j = 0; j < 32; j += 8)
    wT[(size_t)(bx + ty + j) * K + by + tx] = (bf16)tile[tx][ty + j];
}

// ---------------- fused gates GEMM: alpha/beta ----------------
// Block tile 128(M) x 64(N); computes both ih and gate, shares the A tile.
// Double-buffered LDS + async global->LDS loads.
__global__ __launch_bounds__(256) void k_gemm_gates(
    const bf16* __restrict__ xb, const bf16* __restrict__ wihT,
    const bf16* __restrict__ wgT, const float* __restrict__ b_ih,
    const float* __restrict__ b_gate, const float* __restrict__ log_a,
    float* __restrict__ alpha, float* __restrict__ beta) {
  __shared__ bf16 smA[2][128 * 40];
  __shared__ bf16 smBi[2][64 * 40];
  __shared__ bf16 smBg[2][64 * 40];
  const int tid = threadIdx.x;
  const int lane = tid & 31, w = tid >> 5;
  const int wm = w & 3, wn = w >> 2;         // 4x2 waves, each 32x32
  const int bm = blockIdx.y * 128;
  const int bn = blockIdx.x * 64;
  v8f acc_i[2][2] = {};
  v8f acc_g[2][2] = {};

  auto issue = [&](int buf, int kt) {
    #pragma unroll
    for (int j = 0; j < 2; ++j) {
      const int q = tid + j * 256;
      const int r = q >> 2, c = (q & 3) << 3;
      cp16_g2l(xb + (size_t)(bm + r) * DM + kt + c, &smA[buf][r * 40 + c]);
    }
    const int r = tid >> 2, c = (tid & 3) << 3;
    cp16_g2l(wihT + (size_t)(bn + r) * DM + kt + c, &smBi[buf][r * 40 + c]);
    cp16_g2l(wgT  + (size_t)(bn + r) * DM + kt + c, &smBg[buf][r * 40 + c]);
  };

  issue(0, 0);
  for (int i = 0; i < DM / 32; ++i) {
    const int cur = i & 1;
    async_wait_all();     // own wave's loads for buffer `cur` complete
    __syncthreads();      // -> all waves' loads complete; prev reads of other buf done
    if (i + 1 < DM / 32) issue(cur ^ 1, (i + 1) * 32);

    v16bf af[2], bif[2], bgf[2];
    #pragma unroll
    for (int mi = 0; mi < 2; ++mi)
      af[mi] = frag_a(smA[cur], 40, wm * 32 + mi * 16 + (lane & 15), lane);
    #pragma unroll
    for (int ni = 0; ni < 2; ++ni) {
      bif[ni] = frag_b(smBi[cur], 40, wn * 32 + ni * 16 + (lane & 15), lane);
      bgf[ni] = frag_b(smBg[cur], 40, wn * 32 + ni * 16 + (lane & 15), lane);
    }
    #pragma unroll
    for (int mi = 0; mi < 2; ++mi)
      #pragma unroll
      for (int ni = 0; ni < 2; ++ni) {
        acc_i[mi][ni] = WMMA_BF16(af[mi], bif[ni], acc_i[mi][ni]);
        acc_g[mi][ni] = WMMA_BF16(af[mi], bgf[ni], acc_g[mi][ni]);
      }
  }

  const int nlo = lane & 15;
  const int rof = (lane < 16) ? 0 : 8;
  #pragma unroll
  for (int mi = 0; mi < 2; ++mi)
    #pragma unroll
    for (int ni = 0; ni < 2; ++ni) {
      const int col = bn + wn * 32 + ni * 16 + nlo;
      const float bi = b_ih[col];
      const float bg = b_gate[col];
      const float ae = __expf(log_a[col]);
      #pragma unroll
      for (int r = 0; r < 8; ++r) {
        const int row = bm + wm * 32 + mi * 16 + rof + r;
        const float ih = acc_i[mi][ni][r] + bi;
        const float g = 1.0f / (1.0f + __expf(-(acc_g[mi][ni][r] + bg)));
        const size_t idx = (size_t)row * DH + col;
        alpha[idx] = (1.0f - g) * ae;
        beta[idx] = g * ih;
      }
    }
}

// ---------------- 3-phase chunked scan ----------------
__global__ __launch_bounds__(256) void k_scan_p1(const float* __restrict__ alpha,
                                                 const float* __restrict__ beta,
                                                 float* __restrict__ cA,
                                                 float* __restrict__ cB) {
  const int gid = blockIdx.x * 256 + threadIdx.x;   // < BB*NC*DH
  const int h = gid % DH;
  const int bc = gid / DH;
  const int c = bc % NC;
  const int b = bc / NC;
  const size_t base = ((size_t)b * TT + (size_t)c * CL) * DH + h;
  float A = 1.0f, Bv = 0.0f;
  #pragma unroll 4
  for (int i = 0; i < CL; ++i) {
    const float a = alpha[base + (size_t)i * DH];
    const float bb = beta[base + (size_t)i * DH];
    Bv = a * Bv + bb;
    A *= a;
  }
  const int ch = b * DH + h;
  cA[(size_t)c * (BB * DH) + ch] = A;
  cB[(size_t)c * (BB * DH) + ch] = Bv;
}

__global__ __launch_bounds__(256) void k_scan_p2(const float* __restrict__ cA,
                                                 const float* __restrict__ cB,
                                                 float* __restrict__ cIn) {
  const int ch = blockIdx.x * 256 + threadIdx.x;    // < BB*DH
  float hprev = 0.0f;
  #pragma unroll
  for (int c = 0; c < NC; ++c) {
    const size_t idx = (size_t)c * (BB * DH) + ch;
    cIn[idx] = hprev;
    hprev = cA[idx] * hprev + cB[idx];
  }
}

__global__ __launch_bounds__(256) void k_scan_p3(const float* __restrict__ alpha,
                                                 const float* __restrict__ beta,
                                                 const float* __restrict__ cIn,
                                                 float* __restrict__ hout) {
  const int gid = blockIdx.x * 256 + threadIdx.x;
  const int h = gid % DH;
  const int bc = gid / DH;
  const int c = bc % NC;
  const int b = bc / NC;
  const size_t base = ((size_t)b * TT + (size_t)c * CL) * DH + h;
  const int ch = b * DH + h;
  float h0 = cIn[(size_t)c * (BB * DH) + ch];
  #pragma unroll 4
  for (int i = 0; i < CL; ++i) {
    const size_t idx = base + (size_t)i * DH;
    h0 = alpha[idx] * h0 + beta[idx];
    hout[idx] = h0;   // in-place over beta buffer: read-before-write per thread
  }
}

// ---------------- LayerNorm -> bf16 ----------------
__global__ __launch_bounds__(256) void k_ln(const float* __restrict__ h,
                                            const float* __restrict__ g,
                                            const float* __restrict__ bt,
                                            bf16* __restrict__ hn) {
  __shared__ float sred[16];
  const size_t base = (size_t)blockIdx.x * DH;
  float lv[4];
  float s = 0.0f, s2 = 0.0f;
  #pragma unroll
  for (int j = 0; j < 4; ++j) {
    lv[j] = h[base + threadIdx.x + j * 256];
    s += lv[j];
    s2 += lv[j] * lv[j];
  }
  #pragma unroll
  for (int m = 16; m >= 1; m >>= 1) {
    s += __shfl_xor(s, m, 32);
    s2 += __shfl_xor(s2, m, 32);
  }
  const int w = threadIdx.x >> 5;
  if ((threadIdx.x & 31) == 0) { sred[w] = s; sred[8 + w] = s2; }
  __syncthreads();
  if (threadIdx.x == 0) {
    float a = 0.0f, b2 = 0.0f;
    #pragma unroll
    for (int i = 0; i < 8; ++i) { a += sred[i]; b2 += sred[8 + i]; }
    sred[0] = a; sred[8] = b2;
  }
  __syncthreads();
  const float mu = sred[0] * (1.0f / DH);
  const float var = sred[8] * (1.0f / DH) - mu * mu;
  const float rs = rsqrtf(var + LN_EPS);
  #pragma unroll
  for (int j = 0; j < 4; ++j) {
    const int i = threadIdx.x + j * 256;
    hn[base + i] = (bf16)((lv[j] - mu) * rs * g[i] + bt[i]);
  }
}

// ---------------- generic WMMA GEMM (K=1024, N=1024) ----------------
// Double-buffered LDS + async global->LDS loads.
template <int STORE_BF16>
__global__ __launch_bounds__(256) void k_gemm(const bf16* __restrict__ A,
                                              const bf16* __restrict__ BT,
                                              const float* __restrict__ bias,
                                              void* __restrict__ outp) {
  __shared__ bf16 smA[2][128 * 40];
  __shared__ bf16 smB[2][128 * 40];
  const int tid = threadIdx.x;
  const int lane = tid & 31, w = tid >> 5;
  const int wm = w & 3, wn = w >> 2;          // 4x2 waves, each 32x64
  const int bm = blockIdx.y * 128;
  const int bn = blockIdx.x * 128;
  v8f acc[2][4] = {};

  auto issue = [&](int buf, int kt) {
    #pragma unroll
    for (int j = 0; j < 2; ++j) {
      const int q = tid + j * 256;
      const int r = q >> 2, c = (q & 3) << 3;
      cp16_g2l(A  + (size_t)(bm + r) * 1024 + kt + c, &smA[buf][r * 40 + c]);
      cp16_g2l(BT + (size_t)(bn + r) * 1024 + kt + c, &smB[buf][r * 40 + c]);
    }
  };

  issue(0, 0);
  for (int i = 0; i < 32; ++i) {
    const int cur = i & 1;
    async_wait_all();
    __syncthreads();
    if (i + 1 < 32) issue(cur ^ 1, (i + 1) * 32);

    v16bf af[2], bfr[4];
    #pragma unroll
    for (int mi = 0; mi < 2; ++mi)
      af[mi] = frag_a(smA[cur], 40, wm * 32 + mi * 16 + (lane & 15), lane);
    #pragma unroll
    for (int ni = 0; ni < 4; ++ni)
      bfr[ni] = frag_b(smB[cur], 40, wn * 64 + ni * 16 + (lane & 15), lane);
    #pragma unroll
    for (int mi = 0; mi < 2; ++mi)
      #pragma unroll
      for (int ni = 0; ni < 4; ++ni)
        acc[mi][ni] = WMMA_BF16(af[mi], bfr[ni], acc[mi][ni]);
  }

  const int nlo = lane & 15;
  const int rof = (lane < 16) ? 0 : 8;
  #pragma unroll
  for (int mi = 0; mi < 2; ++mi)
    #pragma unroll
    for (int ni = 0; ni < 4; ++ni) {
      const int col = bn + wn * 64 + ni * 16 + nlo;
      const float bs = bias[col];
      #pragma unroll
      for (int r = 0; r < 8; ++r) {
        const int row = bm + wm * 32 + mi * 16 + rof + r;
        const float v = acc[mi][ni][r] + bs;
        const size_t idx = (size_t)row * 1024 + col;
        if (STORE_BF16) ((bf16*)outp)[idx] = (bf16)v;
        else            ((float*)outp)[idx] = v;
      }
    }
}

// ---------------- host ----------------
extern "C" void kernel_launch(void* const* d_in, const int* in_sizes, int n_in,
                              void* d_out, int out_size, void* d_ws, size_t ws_size,
                              hipStream_t stream) {
  const float* x      = (const float*)d_in[0];
  const float* W_ih   = (const float*)d_in[1];
  const float* b_ih   = (const float*)d_in[2];
  const float* W_gate = (const float*)d_in[3];
  const float* b_gate = (const float*)d_in[4];
  const float* log_a  = (const float*)d_in[5];
  const float* ln_g   = (const float*)d_in[6];
  const float* ln_b   = (const float*)d_in[7];
  const float* W_out  = (const float*)d_in[8];
  const float* b_out  = (const float*)d_in[9];
  const float* W_proj = (const float*)d_in[10];
  const float* b_proj = (const float*)d_in[11];

  char* ws = (char*)d_ws;
  size_t off = 0;
  auto alloc = [&](size_t bytes) -> void* {
    void* p = ws + off;
    off += (bytes + 255) & ~(size_t)255;
    return p;
  };
  bf16*  xb     = (bf16*)alloc((size_t)MTOT * DM * sizeof(bf16));
  bf16*  wihT   = (bf16*)alloc((size_t)DH * DM * sizeof(bf16));
  bf16*  wgT    = (bf16*)alloc((size_t)DH * DM * sizeof(bf16));
  bf16*  woutT  = (bf16*)alloc((size_t)DH * DH * sizeof(bf16));
  bf16*  wprojT = (bf16*)alloc((size_t)DM * DH * sizeof(bf16));
  float* alpha  = (float*)alloc((size_t)MTOT * DH * sizeof(float));
  float* beta   = (float*)alloc((size_t)MTOT * DH * sizeof(float));
  float* cA     = (float*)alloc((size_t)NC * BB * DH * sizeof(float));
  float* cB     = (float*)alloc((size_t)NC * BB * DH * sizeof(float));
  float* cIn    = (float*)alloc((size_t)NC * BB * DH * sizeof(float));
  bf16*  hn     = (bf16*)alloc((size_t)MTOT * DH * sizeof(bf16));
  bf16*  outb   = (bf16*)alloc((size_t)MTOT * DH * sizeof(bf16));

  // 1. precision conversion (one-time per call)
  k_cvt<<<dim3((MTOT * DM) / 1024), dim3(256), 0, stream>>>(x, xb);
  k_transpose<<<dim3(DH / 32, DM / 32), dim3(32, 8), 0, stream>>>(W_ih, wihT, DM, DH);
  k_transpose<<<dim3(DH / 32, DM / 32), dim3(32, 8), 0, stream>>>(W_gate, wgT, DM, DH);
  k_transpose<<<dim3(DH / 32, DH / 32), dim3(32, 8), 0, stream>>>(W_out, woutT, DH, DH);
  k_transpose<<<dim3(DM / 32, DH / 32), dim3(32, 8), 0, stream>>>(W_proj, wprojT, DH, DM);

  // 2. fused input GEMMs -> alpha, beta
  k_gemm_gates<<<dim3(DH / 64, MTOT / 128), dim3(256), 0, stream>>>(
      xb, wihT, wgT, b_ih, b_gate, log_a, alpha, beta);

  // 3. chunked linear-recurrence scan (h written over beta)
  k_scan_p1<<<dim3((BB * NC * DH) / 256), dim3(256), 0, stream>>>(alpha, beta, cA, cB);
  k_scan_p2<<<dim3((BB * DH) / 256), dim3(256), 0, stream>>>(cA, cB, cIn);
  k_scan_p3<<<dim3((BB * NC * DH) / 256), dim3(256), 0, stream>>>(alpha, beta, cIn, beta);

  // 4. LayerNorm -> hn (bf16)
  k_ln<<<dim3(MTOT), dim3(256), 0, stream>>>(beta, ln_g, ln_b, hn);

  // 5. out = hn @ W_out + b_out (bf16), final = out @ W_proj + b_proj (fp32)
  k_gemm<1><<<dim3(DH / 128, MTOT / 128), dim3(256), 0, stream>>>(hn, woutT, b_out, outb);
  k_gemm<0><<<dim3(DM / 128, MTOT / 128), dim3(256), 0, stream>>>(outb, wprojT, b_proj, d_out);
}
